// CoTLayer_14465449853593
// MI455X (gfx1250) — matrix operand order repaired and checked by reading, not compiled, not run on hardware
//
#include <hip/hip_runtime.h>
#include <hip/hip_bf16.h>
#include <cstdint>

#define B_    8
#define DIM_  64
#define H_    128
#define W_    128
#define HW_   (H_ * W_)
#define EMB_  72
#define EPS_  1e-5f

#define XW    130          // padded x row width (1-col halo each side)
#define XST   66           // ushort stride per (row,col) entry of x tile (64ch + pad)
#define KST   66           // ushort stride per pixel of k storage
#define H1ST  34           // ushort stride per col of embed-hidden storage
#define KWST  162          // ushort stride per row of conv weights (160 + pad)
#define ZOFF  (3 * XW * XST)   // 16-ushort zero block (padding tap 9)

typedef __bf16 v16bf __attribute__((ext_vector_type(16)));
typedef float  v8f   __attribute__((ext_vector_type(8)));
typedef float  v4f   __attribute__((ext_vector_type(4)));

union BF16x16 { unsigned short us[16]; unsigned au[8]; v16bf v; };

static __device__ __forceinline__ unsigned short f2bf(float f) {
  union { float f; unsigned u; } v; v.f = f;
  return (unsigned short)(v.u >> 16);   // truncate: 1 VALU op
}

// ---------------------------------------------------------------------------
// Kernel 1: per-(b,row) front end. WMMA GEMMs for grouped conv, embed MLP, c1.
// Conv K-dim permuted tap-major (k' = tap*16 + ci) so B fragments are
// channel-contiguous dword loads from the pixel-major x tile.
// ---------------------------------------------------------------------------
__global__ __launch_bounds__(256) void cot_front(
    const float* __restrict__ x,     const float* __restrict__ ke_w,
    const float* __restrict__ ke_g,  const float* __restrict__ ke_b,
    const float* __restrict__ em_w1, const float* __restrict__ em_g1,
    const float* __restrict__ em_b1, const float* __restrict__ em_w2,
    const float* __restrict__ em_b2, const float* __restrict__ c1_w,
    const float* __restrict__ c1_g,  const float* __restrict__ c1_b,
    float* __restrict__ kout, float* __restrict__ vpre,
    float* __restrict__ wraw, float* __restrict__ gnsums) {
  // x tile: [(row*XW + col)][66] -> 64 channels contiguous + pad; + zero block
  __shared__ unsigned short xall[3 * XW * XST + 16];
  __shared__ unsigned short ksh[W_ * KST];        // k: [pixel][66] ch-contig
  __shared__ unsigned short h1sh[8 * 16 * H1ST];  // hidden: per wave [col][34]
  __shared__ unsigned short kwsh[DIM_ * KWST];    // conv W, tap-major permuted
  __shared__ unsigned short w1sh[32 * 130];
  __shared__ unsigned short w2sh[80 * H1ST];
  __shared__ unsigned short c1sh[64 * 66];
  __shared__ float gnsh[16];

  const int tid  = threadIdx.x;
  const int bb   = blockIdx.x >> 7;
  const int h    = blockIdx.x & 127;
  const int wv   = tid >> 5;
  const int lane = tid & 31;
  const int col  = lane & 15;
  const int wpix = wv * 16 + col;
  const int hi8  = (lane < 16) ? 0 : 8;
  const bool lo16 = (lane < 16);
  const float rs1 = rsqrtf(1.0f + EPS_);

  if (tid < 16) { gnsh[tid] = 0.0f; xall[ZOFF + tid] = 0; }

  // ---- stage x rows (h-1,h,h+1): pixel-major, channel-contiguous, halo=0 ----
  for (int i = tid; i < DIM_ * 3 * XW; i += 256) {
    int c   = i / (3 * XW);
    int rem = i - c * (3 * XW);
    int r   = rem / XW;
    int wc  = rem - r * XW;          // 0..129, global w = wc-1
    int hh  = h + r - 1;
    int gw  = wc - 1;
    float val = 0.0f;
    if (hh >= 0 && hh < H_ && gw >= 0 && gw < W_)
      val = x[(((size_t)bb * DIM_ + c) * H_ + hh) * W_ + gw];
    xall[(r * XW + wc) * XST + c] = f2bf(val);
  }
  // ---- stage weights (bf16; conv permuted k' = tap*16 + ci, tap 9 zero) ----
  for (int i = tid; i < DIM_ * 160; i += 256) {
    int m = i / 160, kp = i - m * 160;
    int tap = kp >> 4, ci = kp & 15;
    float val = (tap < 9) ? ke_w[m * 144 + ci * 9 + tap] : 0.0f;
    kwsh[m * KWST + kp] = f2bf(val);
  }
  for (int i = tid; i < 32 * 128; i += 256)
    w1sh[(i >> 7) * 130 + (i & 127)] = f2bf(em_w1[i]);
  for (int i = tid; i < 80 * 32; i += 256) {
    int m = i >> 5, k = i & 31;
    w2sh[m * H1ST + k] = (m < EMB_) ? f2bf(em_w2[m * 32 + k]) : (unsigned short)0;
  }
  for (int i = tid; i < 64 * 64; i += 256)
    c1sh[(i >> 6) * 66 + (i & 63)] = f2bf(c1_w[i]);
  __syncthreads();

  // A fragment: row (lane&15) of LDS-resident weights, K offset kk.
  auto loadA = [&](const unsigned short* wlds, int rs, int kk) -> v16bf {
    BF16x16 A;
    const unsigned short* row = wlds + (lane & 15) * rs + kk + (lo16 ? 0 : 16);
#pragma unroll
    for (int p = 0; p < 8; ++p) {
      int kb = p << 1;  // 0,2,4,...,14
      A.au[p] = *(const unsigned*)(row + kb);
    }
    return A.v;
  };
  // B fragment: 16 K-contiguous bf16 starting at rp (per-lane pointer).
  auto loadB = [&](const unsigned short* rp) -> v16bf {
    BF16x16 Bm;
#pragma unroll
    for (int p = 0; p < 8; ++p) Bm.au[p] = *(const unsigned*)(rp + 2 * p);
    return Bm.v;
  };

  // ---- grouped 3x3 conv (4 groups, M=16, K=144->160 tap-major) -> k ----
#pragma unroll
  for (int g = 0; g < 4; ++g) {
    v8f acc = {};
#pragma unroll
    for (int kk = 0; kk < 160; kk += 32) {
      v16bf Av = loadA(kwsh + g * 16 * KWST, KWST, kk);
      const int tapLo = kk >> 4, tapHi = tapLo + 1;           // compile-time
      int oLo = ((tapLo / 3) * XW + wpix + (tapLo % 3)) * XST + g * 16;
      int oHi = (tapHi < 9)
                    ? ((tapHi / 3) * XW + wpix + (tapHi % 3)) * XST + g * 16
                    : ZOFF;
      v16bf Bv = loadB(xall + (lo16 ? oLo : oHi));
      acc = __builtin_amdgcn_wmma_f32_16x16x32_bf16(false, Av, false, Bv,
                                                    (short)0, acc, false, false);
    }
#pragma unroll
    for (int r = 0; r < 8; ++r) {
      int ch = g * 16 + r + hi8;
      float val = fmaxf(acc[r] * ke_g[ch] * rs1 + ke_b[ch], 0.0f);
      kout[((size_t)bb * DIM_ + ch) * HW_ + h * W_ + wpix] = val;
      ksh[wpix * KST + ch] = f2bf(val);
    }
  }
  __syncthreads();

  // ---- embed stage 1: em_w1 (32x128) over qk=[x;k], BN+ReLU -> h1 ----
#pragma unroll
  for (int mt = 0; mt < 2; ++mt) {
    v8f acc = {};
#pragma unroll
    for (int kk = 0; kk < 128; kk += 32) {
      const unsigned short* bp =
          (kk < 64) ? xall + (XW + wpix + 1) * XST + kk + (lo16 ? 0 : 16)
                    : ksh + wpix * KST + (kk - 64) + (lo16 ? 0 : 16);
      acc = __builtin_amdgcn_wmma_f32_16x16x32_bf16(
          false, loadA(w1sh + mt * 16 * 130, 130, kk),
          false, loadB(bp), (short)0, acc, false, false);
    }
#pragma unroll
    for (int r = 0; r < 8; ++r) {
      int ch = mt * 16 + r + hi8;
      float val = fmaxf(acc[r] * em_g1[ch] * rs1 + em_b1[ch], 0.0f);
      h1sh[(wv * 16 + col) * H1ST + ch] = f2bf(val);
    }
  }
  __syncthreads();

  // ---- embed stage 2: em_w2 (72x32) + bias -> wraw, GN partial sums ----
#pragma unroll
  for (int mt = 0; mt < 5; ++mt) {
    v8f acc = {};
    acc = __builtin_amdgcn_wmma_f32_16x16x32_bf16(
        false, loadA(w2sh + mt * 16 * H1ST, H1ST, 0),
        false, loadB(h1sh + (wv * 16 + col) * H1ST + (lo16 ? 0 : 16)),
        (short)0, acc, false, false);
#pragma unroll
    for (int r = 0; r < 8; ++r) {
      int ch = mt * 16 + r + hi8;
      if (ch < EMB_) {
        float val = acc[r] + em_b2[ch];
        wraw[((size_t)bb * EMB_ + ch) * HW_ + h * W_ + wpix] = val;
        int gg = ch / 9;
        atomicAdd(&gnsh[gg], val);
        atomicAdd(&gnsh[8 + gg], val * val);
      }
    }
  }

  // ---- value path: c1_w (64x64) + BN -> vpre ----
#pragma unroll
  for (int mt = 0; mt < 4; ++mt) {
    v8f acc = {};
#pragma unroll
    for (int kk = 0; kk < 64; kk += 32)
      acc = __builtin_amdgcn_wmma_f32_16x16x32_bf16(
          false, loadA(c1sh + mt * 16 * 66, 66, kk),
          false, loadB(xall + (XW + wpix + 1) * XST + kk + (lo16 ? 0 : 16)),
          (short)0, acc, false, false);
#pragma unroll
    for (int r = 0; r < 8; ++r) {
      int ch = mt * 16 + r + hi8;
      vpre[((size_t)bb * DIM_ + ch) * HW_ + h * W_ + wpix] =
          acc[r] * c1_g[ch] * rs1 + c1_b[ch];
    }
  }

  __syncthreads();
  if (tid < 16) atomicAdd(&gnsums[bb * 16 + tid], gnsh[tid]);
}

// ---------------------------------------------------------------------------
// Kernel 2: finalize GroupNorm statistics per (batch, group).
// ---------------------------------------------------------------------------
__global__ __launch_bounds__(64) void cot_gnstat(const float* __restrict__ sums,
                                                 float* __restrict__ stat) {
  int i = threadIdx.x;
  if (i >= 64) return;
  int bb = i >> 3, g = i & 7;
  float n  = (float)(9 * HW_);
  float s  = sums[bb * 16 + g];
  float ss = sums[bb * 16 + 8 + g];
  float mu = s / n;
  float var = ss / n - mu * mu;
  stat[bb * 16 + g]     = mu;
  stat[bb * 16 + 8 + g] = rsqrtf(var + EPS_);
}

// ---------------------------------------------------------------------------
// Kernel 3: GN-normalize w, 3x3 dynamic aggregation, BN+SiLU, SE gap sums.
// Block = 2 image rows. vpre rows h0-1..h0+2 staged to LDS with
// global_load_async_to_lds_b128 (ASYNCcnt); vertical OOB rows pre-zeroed,
// horizontal bounds folded once per pixel into the 9 normalized weights.
// ---------------------------------------------------------------------------
__global__ __launch_bounds__(256) void cot_agg(
    const float* __restrict__ wraw, const float* __restrict__ gnstat,
    const float* __restrict__ em_gn_g, const float* __restrict__ em_gn_b,
    const float* __restrict__ vpre, const float* __restrict__ kin,
    const float* __restrict__ bn_g, const float* __restrict__ bn_b,
    float* __restrict__ vout, float* __restrict__ gap) {
  __shared__ float vsh[DIM_ * 4 * W_];   // [ch][4 rows][128 w] fp32 = 128 KB
  __shared__ float gacc[64];
  const int tid = threadIdx.x;
  if (tid < 64) gacc[tid] = 0.0f;

  const int p0 = blockIdx.x * 256;       // 256 pixels = 2 rows, one batch
  const int bb = p0 >> 14;
  const int h0 = (p0 & (HW_ - 1)) >> 7;  // even
  const int hw = (p0 & (HW_ - 1)) + tid;
  const int h  = hw >> 7, w = hw & 127;

  // ---- async-stage vpre rows h0-1 .. h0+2 (zero OOB rows) ----
  const unsigned vshBase = (unsigned)(uintptr_t)vsh;
#pragma unroll
  for (int it = 0; it < 32; ++it) {
    int idx = tid + it * 256;            // 0..8191 chunks of 4 floats
    int c   = idx >> 7;
    int rem = idx & 127;
    int r   = rem >> 5;                  // 0..3
    int w4  = (rem & 31) << 2;
    int hh  = h0 - 1 + r;
    unsigned ldsOff = vshBase + (unsigned)(((c * 4 + r) * W_ + w4) * 4);
    if (hh >= 0 && hh < H_) {
      int goff = ((bb * DIM_ + c) * HW_ + hh * W_ + w4) * 4;   // bytes
      asm volatile("global_load_async_to_lds_b128 %0, %1, %2"
                   :: "v"(ldsOff), "v"(goff), "s"(vpre) : "memory");
    } else {
      v4f z = {};
      *(v4f*)((char*)vsh + (ldsOff - vshBase)) = z;
    }
  }
  asm volatile("s_wait_asynccnt 0" ::: "memory");
  __syncthreads();

  const float rs1 = rsqrtf(1.0f + EPS_);
  const int lr = (h & 1) + 1;            // local center row in vsh (1 or 2)
  const int wL = (w > 0) ? w - 1 : 0;    // clamped columns
  const int wR = (w < W_ - 1) ? w + 1 : W_ - 1;
  const float mL = (w > 0) ? 1.0f : 0.0f;
  const float mR = (w < W_ - 1) ? 1.0f : 0.0f;

  for (int cg = 0; cg < 8; ++cg) {       // GN group index == cg (9 ch/group)
    float mu = gnstat[bb * 16 + cg];
    float rs = gnstat[bb * 16 + 8 + cg];
    float wn[9];
#pragma unroll
    for (int tap = 0; tap < 9; ++tap) {
      int j = cg * 9 + tap;
      const int dx = tap % 3 - 1;
      float mask = (dx < 0) ? mL : (dx > 0) ? mR : 1.0f;
      float raw = wraw[((size_t)bb * EMB_ + j) * HW_ + hw];
      wn[tap] = ((raw - mu) * rs * em_gn_g[j] + em_gn_b[j]) * mask;
    }
#pragma unroll
    for (int sp = 0; sp < 8; ++sp) {
      int ch = sp * 8 + cg;
      const float* vp = vsh + (ch * 4 + lr) * W_;   // center row of channel
      float s = 0.0f;
#pragma unroll
      for (int tap = 0; tap < 9; ++tap) {
        const int dy = tap / 3 - 1, dx = tap % 3 - 1;
        const int wc = (dx < 0) ? wL : (dx > 0) ? wR : w;
        s += vp[dy * W_ + wc] * wn[tap];
      }
      float vb  = s * bn_g[ch] * rs1 + bn_b[ch];
      float sig = 1.0f / (1.0f + __expf(-vb));
      float v   = vb * sig;
      vout[((size_t)bb * DIM_ + ch) * HW_ + hw] = v;
      float kk = kin[((size_t)bb * DIM_ + ch) * HW_ + hw];
      atomicAdd(&gacc[ch], v + kk);
    }
  }
  __syncthreads();
  if (tid < 64) atomicAdd(&gap[bb * 64 + tid], gacc[tid]);
}

// ---------------------------------------------------------------------------
// Kernel 4: SE MLP + radix-2 softmax per batch.
// ---------------------------------------------------------------------------
__global__ __launch_bounds__(64) void cot_se(
    const float* __restrict__ gap,  const float* __restrict__ se_w1,
    const float* __restrict__ se_b1, const float* __restrict__ se_g,
    const float* __restrict__ se_bb, const float* __restrict__ se_w2,
    const float* __restrict__ se_b2, float* __restrict__ att) {
  __shared__ float a1[32];
  __shared__ float gm[64];
  int bb = blockIdx.x, t = threadIdx.x;
  gm[t] = gap[bb * 64 + t] * (1.0f / (float)HW_);
  __syncthreads();
  const float rs1 = rsqrtf(1.0f + EPS_);
  if (t < 32) {
    float s = se_b1[t];
    for (int c = 0; c < 64; ++c) s += se_w1[t * 64 + c] * gm[c];
    s = s * se_g[t] * rs1 + se_bb[t];
    a1[t] = fmaxf(s, 0.0f);
  }
  __syncthreads();
  float z0 = se_b2[2 * t], z1 = se_b2[2 * t + 1];
  for (int j = 0; j < 32; ++j) {
    z0 += se_w2[(2 * t) * 32 + j] * a1[j];
    z1 += se_w2[(2 * t + 1) * 32 + j] * a1[j];
  }
  float m  = fmaxf(z0, z1);
  float e0 = __expf(z0 - m), e1 = __expf(z1 - m);
  float inv = 1.0f / (e0 + e1);
  att[bb * 128 + 2 * t]     = e0 * inv;
  att[bb * 128 + 2 * t + 1] = e1 * inv;
}

// ---------------------------------------------------------------------------
// Kernel 5: out = v * a0 + k * a1 (streaming, memory bound).
// ---------------------------------------------------------------------------
__global__ __launch_bounds__(256) void cot_gate(
    const float* __restrict__ v, const float* __restrict__ k,
    const float* __restrict__ att, float* __restrict__ out) {
  size_t i  = (size_t)blockIdx.x * 256 + threadIdx.x;
  int plane = (int)(i >> 14);            // b*64 + c
  int bb = plane >> 6, c = plane & 63;
  float a0 = att[bb * 128 + 2 * c];
  float a1 = att[bb * 128 + 2 * c + 1];
  out[i] = v[i] * a0 + k[i] * a1;
}

// ---------------------------------------------------------------------------
extern "C" void kernel_launch(void* const* d_in, const int* in_sizes, int n_in,
                              void* d_out, int out_size, void* d_ws, size_t ws_size,
                              hipStream_t stream) {
  (void)in_sizes; (void)n_in; (void)out_size; (void)ws_size;
  const float* x       = (const float*)d_in[0];
  const float* ke_w    = (const float*)d_in[1];
  const float* ke_g    = (const float*)d_in[2];
  const float* ke_b    = (const float*)d_in[3];
  const float* em_w1   = (const float*)d_in[4];
  const float* em_g1   = (const float*)d_in[5];
  const float* em_b1   = (const float*)d_in[6];
  const float* em_w2   = (const float*)d_in[7];
  const float* em_b2   = (const float*)d_in[8];
  const float* em_gn_g = (const float*)d_in[9];
  const float* em_gn_b = (const float*)d_in[10];
  const float* c1_w    = (const float*)d_in[11];
  const float* c1_g    = (const float*)d_in[12];
  const float* c1_b    = (const float*)d_in[13];
  const float* bn_g    = (const float*)d_in[14];
  const float* bn_b    = (const float*)d_in[15];
  const float* se_w1   = (const float*)d_in[16];
  const float* se_b1   = (const float*)d_in[17];
  const float* se_g    = (const float*)d_in[18];
  const float* se_bb   = (const float*)d_in[19];
  const float* se_w2   = (const float*)d_in[20];
  const float* se_b2   = (const float*)d_in[21];

  char* ws = (char*)d_ws;
  const size_t PLANE = (size_t)B_ * DIM_ * HW_ * sizeof(float);   // 32 MB
  float* kbuf   = (float*)(ws);
  float* vpre   = (float*)(ws + PLANE);
  float* vbuf   = (float*)(ws + 2 * PLANE);
  float* wraw   = (float*)(ws + 3 * PLANE);                        // 36 MB
  float* gnsums = (float*)(ws + 3 * PLANE + (size_t)B_ * EMB_ * HW_ * sizeof(float));
  float* gnstat = gnsums + B_ * 16;
  float* gap    = gnstat + B_ * 16;
  float* att    = gap + B_ * 64;

  hipMemsetAsync(gnsums, 0, (size_t)(B_ * 16 + B_ * 16 + B_ * 64) * sizeof(float),
                 stream);

  cot_front<<<dim3(B_ * H_), dim3(256), 0, stream>>>(
      x, ke_w, ke_g, ke_b, em_w1, em_g1, em_b1, em_w2, em_b2,
      c1_w, c1_g, c1_b, kbuf, vpre, wraw, gnsums);

  cot_gnstat<<<dim3(1), dim3(64), 0, stream>>>(gnsums, gnstat);

  cot_agg<<<dim3(B_ * HW_ / 256), dim3(256), 0, stream>>>(
      wraw, gnstat, em_gn_g, em_gn_b, vpre, kbuf, bn_g, bn_b, vbuf, gap);

  cot_se<<<dim3(B_), dim3(64), 0, stream>>>(
      gap, se_w1, se_b1, se_g, se_bb, se_w2, se_b2, att);

  cot_gate<<<dim3((B_ * DIM_ * HW_) / 256), dim3(256), 0, stream>>>(
      vbuf, kbuf, att, (float*)d_out);
}
